// NCA_16140487098897
// MI455X (gfx1250) — matrix operand order repaired and checked by reading, not compiled, not run on hardware
//
#include <hip/hip_runtime.h>
#include <hip/hip_bf16.h>

// ---------------------------------------------------------------------------
// NCA step for MI455X (gfx1250, wave32):
//   perceive (depthwise 3x3, 5 fixed stencils) -> 80->64->32->16 MLP via
//   v_wmma_f32_16x16x32_f16 (f32 accumulate) -> masked residual update.
// One wave owns a 16-pixel tile (M dim). Activations staged in wave-private
// LDS; weights staged once per block in LDS in B-fragment layout (f16,
// transposed, K'-padded) so every B fragment is two ds_load_b128.
// Swish uses v_exp_f32 + v_rcp_f32 (fast hw recip), not IEEE division.
// ---------------------------------------------------------------------------

typedef __attribute__((ext_vector_type(16))) _Float16    v16h;
typedef __attribute__((ext_vector_type(8)))  float        v8f;
typedef __attribute__((ext_vector_type(4)))  unsigned int v4u;

#define SDIM 512

// LDS layout (halves):
//   [0, 12288)            : 8 waves x (16 rows x 96 K') activation slices
//   [12288, 18432)        : W1t[64 col][96 K']  (K' = 6c+j, j=5 slot zero)
//   [18432, 20480)        : W2t[32 col][64 K]
//   [20480, 20992)        : W3t[16 col][32 K]
#define ACT_OFF  0
#define W1T_OFF  12288
#define W2T_OFF  (12288 + 6144)
#define W3T_OFF  (12288 + 6144 + 2048)
#define LDS_HALVES 20992

union Frag {
  v16h v;
  _Float16 h[16];
  v4u q[2];
};

union PackH {
  _Float16 h[2];
  unsigned int u;
};

__device__ __forceinline__ float swish_f(float z) {
  // z * sigmoid(z) with fast hw transcendentals: v_exp_f32 + v_rcp_f32
  const float e = __expf(-z);
  return z * __builtin_amdgcn_rcpf(1.0f + e);
}

__device__ __forceinline__ void load_bfrag(Frag& B, const _Float16* base) {
  // element e <-> K = base_K + e (e<8), base_K + 16 + (e-8) (e>=8): matches A
  B.q[0] = *reinterpret_cast<const v4u*>(base);
  B.q[1] = *reinterpret_cast<const v4u*>(base + 16);
}

__global__ __launch_bounds__(256) void nca_step_kernel(
    const float* __restrict__ x, const float* __restrict__ noise,
    const float* __restrict__ W1, const float* __restrict__ b1,
    const float* __restrict__ W2, const float* __restrict__ b2,
    const float* __restrict__ W3, const float* __restrict__ b3,
    float* __restrict__ out)
{
  __shared__ __align__(16) _Float16 smem[LDS_HALVES];

  const int tid   = threadIdx.x;
  const int lane  = tid & 31;
  const int wave  = tid >> 5;
  _Float16* sl    = &smem[ACT_OFF + wave * 1536];
  _Float16* W1t   = &smem[W1T_OFF];
  _Float16* W2t   = &smem[W2T_OFF];
  _Float16* W3t   = &smem[W3T_OFF];

  // ------------- one-time cooperative weight staging (f32 -> f16) -----------
  // W1 pad slots (j == 5): 16 groups x 64 cols
  #pragma unroll
  for (int i = 0; i < 4; ++i) {
    const int idx = tid + i * 256;          // 1024 slots
    const int c5  = idx >> 6;
    const int col = idx & 63;
    W1t[col * 96 + c5 * 6 + 5] = (_Float16)0.0f;
  }
  // W1 [80,64] coalesced read; scatter to W1t[col][6c+j]
  #pragma unroll
  for (int i = 0; i < 20; ++i) {
    const int idx = tid + i * 256;          // 5120 elements
    const int k   = idx >> 6;
    const int col = idx & 63;
    const int c5  = k / 5;
    const int j   = k - 5 * c5;
    W1t[col * 96 + c5 * 6 + j] = (_Float16)W1[idx];
  }
  // W2 [64,32] -> W2t[col][k]
  #pragma unroll
  for (int i = 0; i < 8; ++i) {
    const int idx = tid + i * 256;          // 2048 elements
    const int k   = idx >> 5;
    const int col = idx & 31;
    W2t[col * 64 + k] = (_Float16)W2[idx];
  }
  // W3 [32,16] -> W3t[col][k]
  {
    const int idx = tid;                    // 512 elements, 2 per thread
    const int k0  = idx >> 4;
    const int c0w = idx & 15;
    W3t[c0w * 32 + k0] = (_Float16)W3[idx];
    const int idx2 = tid + 256;
    const int k1   = idx2 >> 4;
    const int c1w  = idx2 & 15;
    W3t[c1w * 32 + k1] = (_Float16)W3[idx2];
  }
  __syncthreads();

  const int bimg  = blockIdx.z;
  const int r     = blockIdx.y;
  const int c0    = blockIdx.x * 128 + wave * 16;  // column base for this wave

  const int ch    = lane & 15;       // perceive channel
  const int half  = lane >> 4;       // 0: lanes 0-15, 1: lanes 16-31
  const int klane = half * 8;        // A / C / D lane-half offset (ISA 7.12.2)
  const int kbB   = half * 8;        // B lane-half K offset (same as A)
  const int n     = lane & 15;       // B column, C/D N index

  const size_t imgBase = (size_t)bimg * SDIM * SDIM;

  // ------------------- perceive: depthwise 3x3, 5 stencils ------------------
  // y stored as [pixel][K'] with K' = 6*c + j (j=0..4 valid, j=5 zero pad).
  #pragma unroll
  for (int it = 0; it < 8; ++it) {
    const int pix = it * 2 + half;
    const int col = c0 + pix;
    float t[3][3];
    #pragma unroll
    for (int dy = 0; dy < 3; ++dy) {
      const int rr  = r + dy - 1;
      const int rrc = min(max(rr, 0), SDIM - 1);
      #pragma unroll
      for (int dxk = 0; dxk < 3; ++dxk) {
        const int cc  = col + dxk - 1;
        const int ccc = min(max(cc, 0), SDIM - 1);
        // unconditional (clamped) load, zero-select for SAME padding
        const float v = x[((imgBase + (size_t)rrc * SDIM + ccc) << 4) + ch];
        t[dy][dxk] = ((rr == rrc) && (cc == ccc)) ? v : 0.0f;
      }
    }
    const float vI  = t[1][1];
    const float gx  = (t[0][2] - t[0][0] + 2.0f*(t[1][2] - t[1][0]) + t[2][2] - t[2][0]) * 0.125f;
    const float gy  = (t[2][0] - t[0][0] + 2.0f*(t[2][1] - t[0][1]) + t[2][2] - t[0][2]) * 0.125f;
    const float lap = 0.25f*(t[0][0] + t[0][2] + t[2][0] + t[2][2])
                    + 0.5f *(t[0][1] + t[1][0] + t[1][2] + t[2][1])
                    - 3.0f * t[1][1];
    const float av  = (t[0][0]+t[0][1]+t[0][2]+t[1][0]+t[1][1]+t[1][2]
                      +t[2][0]+t[2][1]+t[2][2]) * (1.0f/9.0f);

    _Float16* dst = &sl[pix * 96 + ch * 6];   // 4-byte aligned (ch*12 bytes)
    PackH p0; p0.h[0] = (_Float16)vI; p0.h[1] = (_Float16)gx;
    PackH p1; p1.h[0] = (_Float16)gy; p1.h[1] = (_Float16)lap;
    PackH p2; p2.h[0] = (_Float16)av; p2.h[1] = (_Float16)0.0f;
    *reinterpret_cast<unsigned int*>(dst + 0) = p0.u;
    *reinterpret_cast<unsigned int*>(dst + 2) = p1.u;
    *reinterpret_cast<unsigned int*>(dst + 4) = p2.u;
  }
  asm volatile("s_wait_dscnt 0x0" ::: "memory");

  // ---------------- A fragments for layer 1 (16 x 96 f16) -------------------
  const int row = lane & 15;
  Frag A1[3];
  #pragma unroll
  for (int t = 0; t < 3; ++t) {
    A1[t].q[0] = *reinterpret_cast<const v4u*>(&sl[row * 96 + t * 32 + klane]);
    A1[t].q[1] = *reinterpret_cast<const v4u*>(&sl[row * 96 + t * 32 + klane + 16]);
  }

  // ---------------- layer 1: [16x96] x [96x64] + b1, swish ------------------
  Frag B;
  #pragma unroll
  for (int nt = 0; nt < 4; ++nt) {
    v8f acc = {0.0f, 0.0f, 0.0f, 0.0f, 0.0f, 0.0f, 0.0f, 0.0f};
    #pragma unroll
    for (int t = 0; t < 3; ++t) {
      load_bfrag(B, &W1t[(nt * 16 + n) * 96 + t * 32 + kbB]);
      acc = __builtin_amdgcn_wmma_f32_16x16x32_f16(
          false, A1[t].v, false, B.v, (short)0, acc, false, false);
    }
    const float bias = b1[nt * 16 + n];
    #pragma unroll
    for (int i = 0; i < 8; ++i) {
      const float hv = swish_f(acc[i] + bias);
      sl[(i + klane) * 96 + nt * 16 + n] = (_Float16)hv;   // h1[M][N]
    }
  }
  asm volatile("s_wait_dscnt 0x0" ::: "memory");

  // ---------------- layer 2: [16x64] x [64x32] + b2, swish ------------------
  Frag A2[2];
  #pragma unroll
  for (int t = 0; t < 2; ++t) {
    A2[t].q[0] = *reinterpret_cast<const v4u*>(&sl[row * 96 + t * 32 + klane]);
    A2[t].q[1] = *reinterpret_cast<const v4u*>(&sl[row * 96 + t * 32 + klane + 16]);
  }
  #pragma unroll
  for (int nt = 0; nt < 2; ++nt) {
    v8f acc = {0.0f, 0.0f, 0.0f, 0.0f, 0.0f, 0.0f, 0.0f, 0.0f};
    #pragma unroll
    for (int t = 0; t < 2; ++t) {
      load_bfrag(B, &W2t[(nt * 16 + n) * 64 + t * 32 + kbB]);
      acc = __builtin_amdgcn_wmma_f32_16x16x32_f16(
          false, A2[t].v, false, B.v, (short)0, acc, false, false);
    }
    const float bias = b2[nt * 16 + n];
    #pragma unroll
    for (int i = 0; i < 8; ++i) {
      const float hv = swish_f(acc[i] + bias);
      sl[(i + klane) * 96 + nt * 16 + n] = (_Float16)hv;   // h2[M][N]
    }
  }
  asm volatile("s_wait_dscnt 0x0" ::: "memory");

  // ---------------- layer 3: [16x32] x [32x16] + b3 --------------------------
  Frag A3;
  A3.q[0] = *reinterpret_cast<const v4u*>(&sl[row * 96 + klane]);
  A3.q[1] = *reinterpret_cast<const v4u*>(&sl[row * 96 + klane + 16]);
  load_bfrag(B, &W3t[n * 32 + kbB]);
  v8f acc3 = {0.0f, 0.0f, 0.0f, 0.0f, 0.0f, 0.0f, 0.0f, 0.0f};
  acc3 = __builtin_amdgcn_wmma_f32_16x16x32_f16(
      false, A3.v, false, B.v, (short)0, acc3, false, false);

  // ---------------- epilogue: masked residual update -------------------------
  const float bias3 = b3[n];
  #pragma unroll
  for (int i = 0; i < 8; ++i) {
    const int col = c0 + i + klane;                 // pixel M = i + klane
    const size_t pidx = imgBase + (size_t)r * SDIM + col;
    const float mk = (noise[pidx] <= 0.5f) ? 1.0f : 0.0f;
    const size_t xi = (pidx << 4) + n;              // channel = n
    out[xi] = x[xi] + (acc3[i] + bias3) * mk;       // STEP_SIZE = 1.0
  }
}

extern "C" void kernel_launch(void* const* d_in, const int* in_sizes, int n_in,
                              void* d_out, int out_size, void* d_ws, size_t ws_size,
                              hipStream_t stream) {
  (void)in_sizes; (void)n_in; (void)out_size; (void)d_ws; (void)ws_size;
  const float* x     = (const float*)d_in[0];
  const float* noise = (const float*)d_in[1];
  const float* W1    = (const float*)d_in[2];
  const float* b1    = (const float*)d_in[3];
  const float* W2    = (const float*)d_in[4];
  const float* b2    = (const float*)d_in[5];
  const float* W3    = (const float*)d_in[6];
  const float* b3    = (const float*)d_in[7];
  float* out         = (float*)d_out;

  dim3 block(256);                       // 8 waves; each wave: 16 pixels
  dim3 grid(SDIM / 128, SDIM, 4);        // cols/128, rows, batch
  nca_step_kernel<<<grid, block, 0, stream>>>(x, noise, W1, b1, W2, b2, W3, b3, out);
}
